// SelfAttention_59176059404467
// MI455X (gfx1250) — matrix operand order repaired and checked by reading, not compiled, not run on hardware
//
#include <hip/hip_runtime.h>

// ---------------------------------------------------------------------------
// Causal self-attention, B=4, S=4096, D=256, fp32 I/O.
// All matmuls via v_wmma_f32_16x16x32_bf16 (wave32, gfx1250).
// K/V tiles double-buffered in LDS, streamed by the Tensor Data Mover and
// pipelined against WMMA compute via TENSORcnt.
// ---------------------------------------------------------------------------

typedef __attribute__((ext_vector_type(16))) __bf16        v16bf;
typedef __attribute__((ext_vector_type(8)))  float         v8f;
typedef __attribute__((ext_vector_type(4)))  float         f32x4;
typedef __attribute__((ext_vector_type(4)))  __bf16        bf16x4;
typedef __attribute__((ext_vector_type(4)))  unsigned int  u32x4;
typedef __attribute__((ext_vector_type(8)))  unsigned int  u32x8;
typedef __attribute__((ext_vector_type(4)))  int           i32x4;
typedef __attribute__((ext_vector_type(8)))  int           i32x8;

#define WMMA_BF16(A, B_, C_) \
  __builtin_amdgcn_wmma_f32_16x16x32_bf16(false, (A), false, (B_), (short)0, (C_), false, false)

constexpr int BATCH = 4;
constexpr int SEQ   = 4096;
constexpr int DIM   = 256;          // == d_k
constexpr int WMAT  = DIM * DIM;
constexpr int QW    = 16;           // queries per wave
constexpr int NWAVE = 4;            // waves per block (128 threads)
constexpr int QB    = QW * NWAVE;   // 64 queries per block
constexpr int KT    = 32;           // keys per LDS tile / iteration

// --- vectorized WMMA fragment loads (16-byte aligned sources) --------------
__device__ __forceinline__ v16bf frag_split(const __bf16* p) {  // A-layout
  u32x8 t;
  t.lo = *(const u32x4*)(p);        // halves 0..7
  t.hi = *(const u32x4*)(p + 16);   // halves 8..15
  return __builtin_bit_cast(v16bf, t);
}
__device__ __forceinline__ v16bf frag_contig(const __bf16* p) { // contiguous k order, used for A and B
  u32x8 t;
  t.lo = *(const u32x4*)(p);
  t.hi = *(const u32x4*)(p + 8);
  return __builtin_bit_cast(v16bf, t);
}

// --- Tensor Data Mover: 2D tile load Global -> LDS (D# per ISA ch.8) -------
__device__ __forceinline__ void tdm_load_2d(unsigned lds_addr, const void* gp,
                                            unsigned tile_d0, unsigned tile_d1,
                                            unsigned long long s0) {
  unsigned long long ga = (unsigned long long)(uintptr_t)gp;
  u32x4 g0;
  g0.x = 0x1u;                                              // count=1 valid desc
  g0.y = lds_addr;                                          // LDS byte address
  g0.z = (unsigned)(ga & 0xFFFFFFFFu);                      // global_addr[31:0]
  g0.w = (unsigned)((ga >> 32) & 0x01FFFFFFu) | (2u << 30); // addr[56:32]|type=2
  i32x8 g1;
  g1[0] = (int)(1u << 16);                                  // data_size=1 (2B)
  g1[1] = (int)((tile_d0 & 0xFFFFu) << 16);                 // tensor_dim0=tile_d0
  g1[2] = (int)((tile_d0 >> 16) | ((tile_d1 & 0xFFFFu) << 16));
  g1[3] = (int)((tile_d1 >> 16) | (tile_d0 << 16));         // | tile_dim0
  g1[4] = (int)(tile_d1 & 0xFFFFu);                         // tile_dim1
  g1[5] = (int)(s0 & 0xFFFFFFFFull);
  g1[6] = (int)((s0 >> 32) & 0xFFFFull);
  g1[7] = 0;
  i32x4 z4 = {0, 0, 0, 0};
#if __clang_major__ >= 23
  i32x8 z8 = {0, 0, 0, 0, 0, 0, 0, 0};
  __builtin_amdgcn_tensor_load_to_lds(g0, g1, z4, z4, z8, 0);
#else
  __builtin_amdgcn_tensor_load_to_lds(g0, g1, z4, z4, 0);
#endif
}

// ---------------------------------------------------------------------------
// Kernel 0: fp32 weights -> bf16 TRANSPOSED (Wt[n][k]).
// ---------------------------------------------------------------------------
__global__ void convert_w(const float* __restrict__ wq,
                          const float* __restrict__ wk,
                          const float* __restrict__ wv,
                          __bf16* __restrict__ wbf) {
  int i = blockIdx.x * blockDim.x + threadIdx.x;   // out index: col*256 + k
  if (i < WMAT) {
    int col = i >> 8, k = i & 255;
    int src = k * DIM + col;
    wbf[i]            = (__bf16)wq[src];
    wbf[WMAT + i]     = (__bf16)wk[src];
    wbf[2 * WMAT + i] = (__bf16)wv[src];
  }
}

// ---------------------------------------------------------------------------
// Kernel 1: fused QKV projection.  B-fragments preloaded (one load clause),
// then an 8-deep WMMA accumulate chain.  V stored transposed per batch.
// ---------------------------------------------------------------------------
__global__ void qkv_proj(const float* __restrict__ x,
                         const __bf16* __restrict__ wbf,
                         __bf16* __restrict__ qkv) {
  __shared__ alignas(16) __bf16 xs[16 * DIM];   // 8 KB
  const int tid  = threadIdx.x;                 // 128 threads
  const int row0 = blockIdx.x * 16;

  for (int i = tid * 4; i < 16 * DIM; i += 128 * 4) {
    f32x4 xv = *(const f32x4*)&x[(size_t)row0 * DIM + i];
    bf16x4 bv;
#pragma unroll
    for (int j = 0; j < 4; ++j) bv[j] = (__bf16)xv[j];
    *(bf16x4*)&xs[i] = bv;
  }
  __syncthreads();

  const int wave = tid >> 5;
  const int lane = tid & 31;
  const int m    = lane & 15;
  const int hi   = lane >> 4;

  v16bf a[8];
#pragma unroll
  for (int kc = 0; kc < 8; ++kc)
    a[kc] = frag_contig(&xs[m * DIM + kc * 32 + hi * 16]);

  for (int t = wave; t < 48; t += NWAVE) {
    const int mat = t >> 4;
    const int nt  = t & 15;
    const int col = nt * 16 + m;
    const __bf16* wt = wbf + mat * WMAT;        // transposed: [col][k]
    v16bf bf[8];
#pragma unroll
    for (int kc = 0; kc < 8; ++kc)              // one big load clause
      bf[kc] = frag_contig(&wt[col * DIM + kc * 32 + hi * 16]);
    v8f acc = {};
#pragma unroll
    for (int kc = 0; kc < 8; ++kc)              // back-to-back WMMA chain
      acc = WMMA_BF16(a[kc], bf[kc], acc);

    if (mat < 2) {
      __bf16* dst = qkv + (size_t)mat * BATCH * SEQ * DIM;
#pragma unroll
      for (int r = 0; r < 8; ++r)
        dst[(size_t)(row0 + r + hi * 8) * DIM + col] = (__bf16)acc[r];
    } else {
      const int batch = row0 / SEQ;
      const int s0    = row0 % SEQ;
      __bf16* dst = qkv + (size_t)2 * BATCH * SEQ * DIM + (size_t)batch * SEQ * DIM;
#pragma unroll
      for (int r = 0; r < 8; ++r)               // contiguous -> b128 store
        dst[(size_t)col * SEQ + s0 + r + hi * 8] = (__bf16)acc[r];
    }
  }
}

// ---------------------------------------------------------------------------
// Kernel 2: flash attention, TDM double-buffered.
// ---------------------------------------------------------------------------
__global__ void flash_attn(const __bf16* __restrict__ qkv,
                           float* __restrict__ out) {
  // [ks0 16K][ks1 16K][vt0 16K][vt1 16K][ps 4K] ; fp32 O-scratch reuses 0..64K
  __shared__ alignas(16) unsigned char smem[69632];
  __bf16* const ks0 = (__bf16*)(smem);
  __bf16* const ks1 = (__bf16*)(smem + 16384);
  __bf16* const vt0 = (__bf16*)(smem + 32768);
  __bf16* const vt1 = (__bf16*)(smem + 49152);
  __bf16* const ps  = (__bf16*)(smem + 65536);

  const int tid  = threadIdx.x;
  const int wave = tid >> 5;
  const int lane = tid & 31;
  const int m    = lane & 15;
  const int hi   = lane >> 4;

  const int blk   = blockIdx.x;               // B * S/QB = 256 blocks
  const int batch = blk / (SEQ / QB);
  const int qb    = blk % (SEQ / QB);
  const size_t base = (size_t)batch * SEQ * DIM;

  const __bf16* Q  = qkv + base;
  const __bf16* K  = qkv + (size_t)1 * BATCH * SEQ * DIM + base;
  const __bf16* Vt = qkv + (size_t)2 * BATCH * SEQ * DIM + base; // [dim][s]

  const int qbase = qb * QB + wave * QW;

  v16bf qa[8];
#pragma unroll
  for (int kc = 0; kc < 8; ++kc)
    qa[kc] = frag_contig(&Q[(size_t)(qbase + m) * DIM + kc * 32 + hi * 16]);

  v8f o[16];
#pragma unroll
  for (int n = 0; n < 16; ++n) o[n] = (v8f){};
  float mrow[8], lrow[8];
#pragma unroll
  for (int r = 0; r < 8; ++r) { mrow[r] = -3.0e38f; lrow[r] = 0.f; }

  // 1/sqrt(256) folded with log2(e): softmax done in exp2 domain.
  const float scale2 = 0.0625f * 1.44269504088896f;
  const int ktmax = (qb * QB + QB - 1) / KT;   // causal bound (inclusive)

  // prologue: stream tile 0 into buffer 0
  if (wave == 0) {
    tdm_load_2d((unsigned)(uintptr_t)(void*)ks0, K, KT * DIM, 1, KT * DIM);
    tdm_load_2d((unsigned)(uintptr_t)(void*)vt0, Vt, KT, DIM, SEQ);
  }

  for (int kt = 0; kt <= ktmax; ++kt) {
    if (wave == 0) {
      if (kt < ktmax) {  // prefetch next tile into the other buffer
        __bf16* kn = ((kt + 1) & 1) ? ks1 : ks0;
        __bf16* vn = ((kt + 1) & 1) ? vt1 : vt0;
        tdm_load_2d((unsigned)(uintptr_t)(void*)kn,
                    K + (size_t)(kt + 1) * KT * DIM, KT * DIM, 1, KT * DIM);
        tdm_load_2d((unsigned)(uintptr_t)(void*)vn,
                    Vt + (size_t)(kt + 1) * KT, KT, DIM, SEQ);
        __builtin_amdgcn_s_wait_tensorcnt(2);   // in-order: tile kt complete
      } else {
        __builtin_amdgcn_s_wait_tensorcnt(0);
      }
    }
    __syncthreads();                            // tile kt visible to all waves
    const __bf16* ksx = (kt & 1) ? ks1 : ks0;
    const __bf16* vtx = (kt & 1) ? vt1 : vt0;

    if (kt * KT <= qbase + QW - 1) {
      // ---- scores: S = Q * K^T (groups of 4: load clause + WMMA chain) ----
      v8f sc[2];
#pragma unroll
      for (int nt = 0; nt < 2; ++nt) {
        v8f accs = {};
#pragma unroll
        for (int g = 0; g < 2; ++g) {
          v16bf bf[4];
#pragma unroll
          for (int u = 0; u < 4; ++u)
            bf[u] = frag_contig(&ksx[(nt * 16 + m) * DIM + (g * 4 + u) * 32 + hi * 16]);
#pragma unroll
          for (int u = 0; u < 4; ++u)
            accs = WMMA_BF16(qa[g * 4 + u], bf[u], accs);
        }
        sc[nt] = accs;
      }

      // ---- causal mask + online softmax (exp2 domain) ----
      float alpha[8];
#pragma unroll
      for (int r = 0; r < 8; ++r) {
        const int qidx = qbase + r + hi * 8;
        float mx = -3.0e38f;
#pragma unroll
        for (int nt = 0; nt < 2; ++nt) {
          const int kidx = kt * KT + nt * 16 + m;
          float v = sc[nt][r] * scale2;
          v = (kidx <= qidx) ? v : -3.0e38f;
          sc[nt][r] = v;
          mx = fmaxf(mx, v);
        }
#pragma unroll
        for (int off = 1; off < 16; off <<= 1)
          mx = fmaxf(mx, __shfl_xor(mx, off, 32));
        const float mnew = fmaxf(mrow[r], mx);
        float sum = 0.f;
#pragma unroll
        for (int nt = 0; nt < 2; ++nt) {
          const float p = exp2f(sc[nt][r] - mnew);
          sc[nt][r] = p;
          sum += p;
        }
#pragma unroll
        for (int off = 1; off < 16; off <<= 1)
          sum += __shfl_xor(sum, off, 32);
        alpha[r] = exp2f(mrow[r] - mnew);
        lrow[r]  = lrow[r] * alpha[r] + sum;
        mrow[r]  = mnew;
      }

      // ---- P: C-layout -> bf16 A-layout via per-wave LDS scratch ----
      __bf16* pw = &ps[wave * 16 * KT];
#pragma unroll
      for (int r = 0; r < 8; ++r)
#pragma unroll
        for (int nt = 0; nt < 2; ++nt)
          pw[(r + hi * 8) * KT + nt * 16 + m] = (__bf16)sc[nt][r];
      asm volatile("s_wait_dscnt 0" ::: "memory");
      const v16bf pa = frag_contig(&pw[m * KT + hi * 16]);

      // ---- O = diag(alpha)*O + P*V (groups of 4) ----
#pragma unroll
      for (int g = 0; g < 4; ++g) {
        v16bf bf[4];
#pragma unroll
        for (int u = 0; u < 4; ++u)
          bf[u] = frag_contig(&vtx[((g * 4 + u) * 16 + m) * KT + hi * 16]);
#pragma unroll
        for (int u = 0; u < 4; ++u) {
          const int nt2 = g * 4 + u;
          v8f acc = o[nt2];
#pragma unroll
          for (int r = 0; r < 8; ++r) acc[r] *= alpha[r];
          o[nt2] = WMMA_BF16(pa, bf[u], acc);
        }
      }
    }
    __syncthreads();   // all consumers of buffer (kt&1) done before reuse
  }

  // ---- epilogue: O/l through LDS transpose -> coalesced b128 stores ----
  float* ow = (float*)(void*)smem + wave * 16 * DIM;   // 16KB per wave
  float rl[8];
#pragma unroll
  for (int r = 0; r < 8; ++r) rl[r] = 1.0f / lrow[r];
#pragma unroll
  for (int nt2 = 0; nt2 < 16; ++nt2)
#pragma unroll
    for (int r = 0; r < 8; ++r)
      ow[(r + hi * 8) * DIM + nt2 * 16 + m] = o[nt2][r] * rl[r];
  asm volatile("s_wait_dscnt 0" ::: "memory");         // per-wave region
  for (int i = lane; i < 16 * DIM / 4; i += 32) {
    const int row = i >> 6;                // 64 float4 per 256-wide row
    const int c4  = (i & 63) << 2;
    *(f32x4*)&out[base + (size_t)(qbase + row) * DIM + c4] =
        *(const f32x4*)&ow[row * DIM + c4];
  }
}

// ---------------------------------------------------------------------------
extern "C" void kernel_launch(void* const* d_in, const int* in_sizes, int n_in,
                              void* d_out, int out_size, void* d_ws, size_t ws_size,
                              hipStream_t stream) {
  const float* x  = (const float*)d_in[0];
  const float* wq = (const float*)d_in[1];
  const float* wk = (const float*)d_in[2];
  const float* wv = (const float*)d_in[3];
  float* out = (float*)d_out;

  __bf16* wbf = (__bf16*)d_ws;
  __bf16* qkv = (__bf16*)((char*)d_ws + (size_t)3 * WMAT * sizeof(__bf16));

  convert_w<<<WMAT / 256, 256, 0, stream>>>(wq, wk, wv, wbf);
  qkv_proj<<<BATCH * SEQ / 16, 128, 0, stream>>>(x, wbf, qkv);
  flash_attn<<<BATCH * SEQ / QB, 128, 0, stream>>>(qkv, out);
}
